// Attention_6906307412174
// MI455X (gfx1250) — matrix operand order repaired and checked by reading, not compiled, not run on hardware
//
#include <hip/hip_runtime.h>
#include <stdint.h>

#define BATCH  4
#define LCTX   2048
#define DMODEL 1024
#define HEADS  16
#define DHEAD  64

typedef __attribute__((ext_vector_type(16))) __bf16 v16bf;
typedef __attribute__((ext_vector_type(8)))  float  v8f;
typedef __attribute__((ext_vector_type(4)))  float  v4f;
typedef __attribute__((ext_vector_type(4)))  __bf16 v4bf;

union FragB16 { v16bf v; uint4 q[2]; unsigned short u[16]; };
union PackBF4 { v4bf v; uint2 u; };

// native RNE f32 -> bf16 (backend uses v_cvt_pk_bf16_f32)
__device__ __forceinline__ unsigned short f32_bf16(float f) {
  __bf16 h = (__bf16)f;
  return __builtin_bit_cast(unsigned short, h);
}
__device__ __forceinline__ uint2 cvt4(v4f x) {
  PackBF4 p;
  p.v = __builtin_convertvector(x, v4bf);
  return p.u;
}
__device__ __forceinline__ v8f wmma_bf16(const FragB16& a, const FragB16& b, v8f c) {
  return __builtin_amdgcn_wmma_f32_16x16x32_bf16(false, a.v, false, b.v,
                                                 (short)0, c, false, false);
}

// --- CDNA5 async global->LDS copy (ASYNCcnt-tracked, bypasses VGPRs) -------
__device__ __forceinline__ unsigned lds_off_u32(const void* p) {
  // generic LDS addresses on amdgcn are {aperture_hi32, lds_offset_lo32}
  return (unsigned)(unsigned long long)(uintptr_t)p;
}
__device__ __forceinline__ void async_copy_b128(void* lds_dst, const void* gsrc) {
  asm volatile("global_load_async_to_lds_b128 %0, %1, off"
               :: "v"(lds_off_u32(lds_dst)), "v"(gsrc)
               : "memory");
}
__device__ __forceinline__ void wait_async0() {
  asm volatile("s_wait_asynccnt 0" ::: "memory");
}

// ---------------------------------------------------------------- GEMM core
// C[128x128] tile: A[M,K] x W[N,K]^T, bf16 double-buffered through LDS.
#define GT_K 32
#define SA_STRIDE 40                 // 32 + 8 pad (ushort); 80B rows, 16B aligned
#define GBUF (128 * SA_STRIDE)       // ushorts per LDS buffer

__device__ __forceinline__ void stage16(unsigned short* d,
                                        v4f x0, v4f x1, v4f x2, v4f x3) {
  const uint2 c0 = cvt4(x0), c1 = cvt4(x1), c2 = cvt4(x2), c3 = cvt4(x3);
  uint4 t;
  t.x = c0.x; t.y = c0.y; t.z = c1.x; t.w = c1.y;
  *(uint4*)d = t;
  t.x = c2.x; t.y = c2.y; t.z = c3.x; t.w = c3.y;
  *(uint4*)(d + 8) = t;
}

__device__ __forceinline__ void gemm_tile_core(
    const float* __restrict__ A, const float* __restrict__ W,
    int m0, int n0, unsigned short* sA, unsigned short* sB, v8f acc[2][4])
{
  const int tid  = threadIdx.x;
  const int lane = tid & 31;
  const int wv   = tid >> 5;
  const int wM   = wv >> 1;       // 0..3
  const int wN   = wv & 1;        // 0..1
  const int hs   = lane >> 4;     // lane half
  const int l16  = lane & 15;
  const int srow = tid >> 1;      // 0..127
  const int sk   = (tid & 1) * 16;

  const float* pa_row = A + (size_t)(m0 + srow) * DMODEL + sk;
  const float* pb_row = W + (size_t)(n0 + srow) * DMODEL + sk;
  unsigned short* da = sA + srow * SA_STRIDE + sk;
  unsigned short* db = sB + srow * SA_STRIDE + sk;

  {
    const v4f* pa = (const v4f*)pa_row;
    const v4f* pb = (const v4f*)pb_row;
    stage16(da, pa[0], pa[1], pa[2], pa[3]);
    stage16(db, pb[0], pb[1], pb[2], pb[3]);
  }
  __syncthreads();

  for (int k0 = 0; k0 < DMODEL; k0 += GT_K) {
    const int curo = ((k0 >> 5) & 1) * GBUF;
    const int nxto = curo ^ GBUF;
    const bool more = (k0 + GT_K) < DMODEL;

    v4f a0, a1, a2, a3, b0, b1, b2, b3;
    if (more) {
      const v4f* pa = (const v4f*)(pa_row + k0 + GT_K);
      const v4f* pb = (const v4f*)(pb_row + k0 + GT_K);
      a0 = pa[0]; a1 = pa[1]; a2 = pa[2]; a3 = pa[3];
      b0 = pb[0]; b1 = pb[1]; b2 = pb[2]; b3 = pb[3];
      if (k0 + 2 * GT_K < DMODEL) {
        __builtin_prefetch(pa_row + k0 + 2 * GT_K, 0, 1);
        __builtin_prefetch(pb_row + k0 + 2 * GT_K, 0, 1);
      }
    }

    FragB16 aF[2], bF[4];
#pragma unroll
    for (int mt = 0; mt < 2; ++mt) {
      // A-frag 16x32: lane = row; k chunks {hs*8, 16+hs*8}
      const unsigned short* p = sA + curo + (wM * 32 + mt * 16 + l16) * SA_STRIDE;
      aF[mt].q[0] = *(const uint4*)(p + hs * 8);
      aF[mt].q[1] = *(const uint4*)(p + 16 + hs * 8);
    }
#pragma unroll
    for (int nt = 0; nt < 4; ++nt) {
      // B-frag 32x16: lane = column; k chunks {hs*8, 16+hs*8}
      const unsigned short* p = sB + curo + (wN * 64 + nt * 16 + l16) * SA_STRIDE;
      bF[nt].q[0] = *(const uint4*)(p + hs * 8);
      bF[nt].q[1] = *(const uint4*)(p + 16 + hs * 8);
    }
#pragma unroll
    for (int mt = 0; mt < 2; ++mt)
#pragma unroll
      for (int nt = 0; nt < 4; ++nt)
        acc[mt][nt] = wmma_bf16(aF[mt], bF[nt], acc[mt][nt]);

    if (more) {
      stage16(da + nxto, a0, a1, a2, a3);
      stage16(db + nxto, b0, b1, b2, b3);
    }
    __syncthreads();
  }
}

// ---------------------------------------------------------------- kernel 1
__global__ void __launch_bounds__(256)
qkv_gemm_kernel(const float* __restrict__ Aq, const float* __restrict__ Ak,
                const float* __restrict__ Av,
                const float* __restrict__ Wq, const float* __restrict__ Wk,
                const float* __restrict__ Wv,
                const float* __restrict__ bq, const float* __restrict__ bk,
                const float* __restrict__ bv,
                unsigned short* __restrict__ qT, unsigned short* __restrict__ kT,
                unsigned short* __restrict__ vT)
{
  __shared__ __align__(16) unsigned short sA[2 * GBUF];
  __shared__ __align__(16) unsigned short sB[2 * GBUF];

  const float* A; const float* W; const float* bias; unsigned short* outT;
  if (blockIdx.z == 0)      { A = Aq; W = Wq; bias = bq; outT = qT; }
  else if (blockIdx.z == 1) { A = Ak; W = Wk; bias = bk; outT = kT; }
  else                      { A = Av; W = Wv; bias = bv; outT = vT; }

  const int m0 = blockIdx.y * 128;
  const int n0 = blockIdx.x * 128;

  v8f acc[2][4];
  v8f zero = {};
#pragma unroll
  for (int i = 0; i < 2; ++i)
#pragma unroll
    for (int j = 0; j < 4; ++j) acc[i][j] = zero;

  gemm_tile_core(A, W, m0, n0, sA, sB, acc);

  const int lane = threadIdx.x & 31;
  const int wv = threadIdx.x >> 5;
  const int wM = wv >> 1, wN = wv & 1;
  const int hs = lane >> 4, l16 = lane & 15;

#pragma unroll
  for (int nt = 0; nt < 4; ++nt) {
    const int n = n0 + wN * 64 + nt * 16 + l16;
    const float bval = bias[n];
    const int h = n >> 6, dh = n & 63;
#pragma unroll
    for (int mt = 0; mt < 2; ++mt) {
#pragma unroll
      for (int r = 0; r < 8; ++r) {
        const int m  = m0 + wM * 32 + mt * 16 + r + 8 * hs;
        const int bb = m >> 11;          // / LCTX
        const int l  = m & (LCTX - 1);
        outT[(((size_t)bb * HEADS + h) * LCTX + l) * DHEAD + dh] =
            f32_bf16(acc[mt][nt][r] + bval);
      }
    }
  }
}

// ---------------------------------------------------------------- kernel 2
// Flash attention: 128 query rows per WG, async double-buffered KV blocks.
#define QBLK 128
#define KBLK 64
#define SQ_STRIDE 72                  // 64 + 8 pad
#define KBUF (KBLK * SQ_STRIDE)
#define VBUF (DHEAD * SQ_STRIDE)

__device__ __forceinline__ void stage_K_async(unsigned short* sKbuf,
                                              const unsigned short* Kbase,
                                              int kb, int tid) {
#pragma unroll
  for (int i = 0; i < 2; ++i) {
    const int idx = tid * 2 + i;             // 0..511 uint4s
    const int row = idx >> 3, c = (idx & 7) * 8;
    async_copy_b128(&sKbuf[row * SQ_STRIDE + c],
                    &Kbase[(size_t)(kb + row) * DHEAD + c]);
  }
}

__device__ __forceinline__ void scatter_vt(unsigned short* sVTbuf, int dh0, int krow,
                                           uint4 v0, uint4 v1) {
  unsigned short tmp[16];
  *(uint4*)&tmp[0] = v0;
  *(uint4*)&tmp[8] = v1;
#pragma unroll
  for (int i = 0; i < 16; ++i)
    sVTbuf[(dh0 + i) * SQ_STRIDE + krow] = tmp[i];
}

__global__ void __launch_bounds__(256)
flash_attn_kernel(const unsigned short* __restrict__ qT,
                  const unsigned short* __restrict__ kT,
                  const unsigned short* __restrict__ vT,
                  const int* __restrict__ mask,
                  float* __restrict__ attnO)
{
  __shared__ __align__(16) unsigned short sQ[QBLK * SQ_STRIDE];   // 18 KB
  __shared__ __align__(16) unsigned short sK[2 * KBUF];           // 18 KB
  __shared__ __align__(16) unsigned short sVT[2 * VBUF];          // 18 KB (V^T)
  __shared__ __align__(16) unsigned short sP[8 * 16 * SQ_STRIDE]; // 18 KB
  __shared__ float sBias[2 * KBLK];

  const int tid  = threadIdx.x;
  const int lane = tid & 31;
  const int wv   = tid >> 5;
  const int hs   = lane >> 4;
  const int l16  = lane & 15;
  const int bh   = blockIdx.y;        // b*H + h
  const int b    = bh >> 4;
  const int h    = bh & 15;
  const int q0   = blockIdx.x * QBLK;

  const unsigned short* Qbase = qT + (size_t)bh * LCTX * DHEAD;
  const unsigned short* Kbase = kT + (size_t)bh * LCTX * DHEAD;
  const unsigned short* Vbase = vT + (size_t)bh * LCTX * DHEAD;

  const int vkrow = tid >> 2;               // 0..63
  const int vdh0  = (tid & 3) * 16;

  // prologue: async-stage Q (128x64) + K block 0; manual transpose of V block 0
#pragma unroll
  for (int i = 0; i < 4; ++i) {
    const int idx = tid * 4 + i;            // 0..1023 uint4s
    const int row = idx >> 3, c = (idx & 7) * 8;
    async_copy_b128(&sQ[row * SQ_STRIDE + c],
                    &Qbase[(size_t)(q0 + row) * DHEAD + c]);
  }
  stage_K_async(sK, Kbase, 0, tid);
  {
    const uint4* src = (const uint4*)&Vbase[(size_t)vkrow * DHEAD + vdh0];
    uint4 v0 = src[0], v1 = src[1];
    scatter_vt(sVT, vdh0, vkrow, v0, v1);
  }
  if (tid < KBLK)
    sBias[tid] = (mask[b * LCTX + tid] == 0) ? -1.0e9f : 0.0f;
  wait_async0();
  __syncthreads();

  FragB16 aQ[2];
#pragma unroll
  for (int ks = 0; ks < 2; ++ks) {
    const unsigned short* p = &sQ[(wv * 16 + l16) * SQ_STRIDE + ks * 32];
    aQ[ks].q[0] = *(const uint4*)(p + hs * 8);
    aQ[ks].q[1] = *(const uint4*)(p + 16 + hs * 8);
  }

  float row_m[8], row_l[8];
  v8f oacc[4];
  v8f zero = {};
#pragma unroll
  for (int r = 0; r < 8; ++r) { row_m[r] = -3.0e38f; row_l[r] = 0.0f; }
#pragma unroll
  for (int t = 0; t < 4; ++t) oacc[t] = zero;

  const float scale = 0.125f;  // 1/sqrt(64)

  int curK = 0, curV = 0, curB = 0;
  for (int kb = 0; kb < LCTX; kb += KBLK) {
    const bool more = (kb + KBLK) < LCTX;
    uint4 nv0, nv1;
    if (more) {
      // fetch next block while this one computes
      stage_K_async(sK + (curK ^ KBUF), Kbase, kb + KBLK, tid);
      const uint4* src =
          (const uint4*)&Vbase[(size_t)(kb + KBLK + vkrow) * DHEAD + vdh0];
      nv0 = src[0]; nv1 = src[1];
      if (tid < KBLK)
        sBias[(curB ^ KBLK) + tid] =
            (mask[b * LCTX + kb + KBLK + tid] == 0) ? -1.0e9f : 0.0f;
    }
    const unsigned short* sKc = sK + curK;
    const unsigned short* sVc = sVT + curV;
    const float* sBc = sBias + curB;

    // S = Q * K^T  (16 x 64 per wave)
    float sv[4][8];
#pragma unroll
    for (int nt = 0; nt < 4; ++nt) {
      FragB16 bK0, bK1;
      const unsigned short* p0 = &sKc[(nt * 16 + l16) * SQ_STRIDE];
      bK0.q[0] = *(const uint4*)(p0 + hs * 8);
      bK0.q[1] = *(const uint4*)(p0 + 16 + hs * 8);
      const unsigned short* p1 = p0 + 32;
      bK1.q[0] = *(const uint4*)(p1 + hs * 8);
      bK1.q[1] = *(const uint4*)(p1 + 16 + hs * 8);
      v8f s = zero;
      s = wmma_bf16(aQ[0], bK0, s);
      s = wmma_bf16(aQ[1], bK1, s);
      const float biasv = sBc[nt * 16 + l16];
#pragma unroll
      for (int r = 0; r < 8; ++r) sv[nt][r] = s[r] * scale + biasv;
    }

    // online softmax (rows live across 16 lanes within a half)
    float corr[8];
#pragma unroll
    for (int r = 0; r < 8; ++r) {
      float mx = fmaxf(fmaxf(sv[0][r], sv[1][r]), fmaxf(sv[2][r], sv[3][r]));
#pragma unroll
      for (int off = 8; off >= 1; off >>= 1)
        mx = fmaxf(mx, __shfl_xor(mx, off, 32));
      const float newm = fmaxf(row_m[r], mx);
      corr[r] = __expf(row_m[r] - newm);
      row_m[r] = newm;
      float psum = 0.0f;
#pragma unroll
      for (int nt = 0; nt < 4; ++nt) {
        const float p = __expf(sv[nt][r] - newm);
        sv[nt][r] = p;
        psum += p;
      }
#pragma unroll
      for (int off = 8; off >= 1; off >>= 1)
        psum += __shfl_xor(psum, off, 32);
      row_l[r] = row_l[r] * corr[r] + psum;
    }
#pragma unroll
    for (int t = 0; t < 4; ++t)
#pragma unroll
      for (int r = 0; r < 8; ++r)
        oacc[t][r] *= corr[r];

    // P: C-layout -> A-layout via per-wave LDS scratch
    unsigned short* myP = &sP[wv * 16 * SQ_STRIDE];
#pragma unroll
    for (int nt = 0; nt < 4; ++nt)
#pragma unroll
      for (int r = 0; r < 8; ++r)
        myP[(r + 8 * hs) * SQ_STRIDE + nt * 16 + l16] = f32_bf16(sv[nt][r]);
    __builtin_amdgcn_wave_barrier();  // same-wave LDS ops are in-order

    FragB16 aP[2];
#pragma unroll
    for (int ks = 0; ks < 2; ++ks) {
      const unsigned short* p = &myP[l16 * SQ_STRIDE + ks * 32];
      aP[ks].q[0] = *(const uint4*)(p + hs * 8);
      aP[ks].q[1] = *(const uint4*)(p + 16 + hs * 8);
    }
    // O += P * V
#pragma unroll
    for (int dt = 0; dt < 4; ++dt) {
      FragB16 bV0, bV1;
      const unsigned short* p0 = &sVc[(dt * 16 + l16) * SQ_STRIDE];
      bV0.q[0] = *(const uint4*)(p0 + hs * 8);
      bV0.q[1] = *(const uint4*)(p0 + 16 + hs * 8);
      const unsigned short* p1 = p0 + 32;
      bV1.q[0] = *(const uint4*)(p1 + hs * 8);
      bV1.q[1] = *(const uint4*)(p1 + 16 + hs * 8);
      oacc[dt] = wmma_bf16(aP[0], bV0, oacc[dt]);
      oacc[dt] = wmma_bf16(aP[1], bV1, oacc[dt]);
    }

    if (more)
      scatter_vt(sVT + (curV ^ VBUF), vdh0, vkrow, nv0, nv1);
    wait_async0();
    __syncthreads();
    curK ^= KBUF; curV ^= VBUF; curB ^= KBLK;
  }

  // write O (f32, (b,l,d) layout)
#pragma unroll
  for (int r = 0; r < 8; ++r) {
    const int qrow = q0 + wv * 16 + r + 8 * hs;
    const float inv = 1.0f / row_l[r];
    float* dst = attnO + ((size_t)b * LCTX + qrow) * DMODEL + h * DHEAD;
#pragma unroll
    for (int dt = 0; dt < 4; ++dt)
      dst[dt * 16 + l16] = oacc[dt][r] * inv;
  }
}

// ---------------------------------------------------------------- kernel 3
__global__ void __launch_bounds__(256)
oproj_gemm_kernel(const float* __restrict__ A, const float* __restrict__ W,
                  const float* __restrict__ bias, const float* __restrict__ resid,
                  float* __restrict__ xout)
{
  __shared__ __align__(16) unsigned short sA[2 * GBUF];
  __shared__ __align__(16) unsigned short sB[2 * GBUF];

  const int m0 = blockIdx.y * 128;
  const int n0 = blockIdx.x * 128;

  v8f acc[2][4];
  v8f zero = {};
#pragma unroll
  for (int i = 0; i < 2; ++i)
#pragma unroll
    for (int j = 0; j < 4; ++j) acc[i][j] = zero;

  gemm_tile_core(A, W, m0, n0, sA, sB, acc);

  const int lane = threadIdx.x & 31;
  const int wv = threadIdx.x >> 5;
  const int wM = wv >> 1, wN = wv & 1;
  const int hs = lane >> 4, l16 = lane & 15;

#pragma unroll
  for (int nt = 0; nt < 4; ++nt) {
    const int n = n0 + wN * 64 + nt * 16 + l16;
    const float bval = bias[n];
#pragma unroll
    for (int mt = 0; mt < 2; ++mt) {
#pragma unroll
      for (int r = 0; r < 8; ++r) {
        const int m = m0 + wM * 32 + mt * 16 + r + 8 * hs;
        const size_t idx = (size_t)m * DMODEL + n;
        xout[idx] = acc[mt][nt][r] + bval + resid[idx];
      }
    }
  }
}

// ---------------------------------------------------------------- kernel 4
__global__ void __launch_bounds__(256)
layernorm_kernel(const float* __restrict__ x, const float* __restrict__ gamma,
                 const float* __restrict__ beta, float* __restrict__ out)
{
  __shared__ float redS[8], redQ[8], stat[2];
  const int row = blockIdx.x;
  const int tid = threadIdx.x;
  const float4 v = ((const float4*)(x + (size_t)row * DMODEL))[tid];
  float s  = v.x + v.y + v.z + v.w;
  float sq = v.x * v.x + v.y * v.y + v.z * v.z + v.w * v.w;
#pragma unroll
  for (int off = 16; off >= 1; off >>= 1) {
    s  += __shfl_xor(s, off, 32);
    sq += __shfl_xor(sq, off, 32);
  }
  if ((tid & 31) == 0) { redS[tid >> 5] = s; redQ[tid >> 5] = sq; }
  __syncthreads();
  if (tid == 0) {
    float ts = 0.0f, tq = 0.0f;
#pragma unroll
    for (int i = 0; i < 8; ++i) { ts += redS[i]; tq += redQ[i]; }
    const float mu  = ts * (1.0f / DMODEL);
    const float var = tq * (1.0f / DMODEL) - mu * mu;
    stat[0] = mu;
    stat[1] = rsqrtf(var + 1e-5f);
  }
  __syncthreads();
  const float mu = stat[0], rstd = stat[1];
  const float4 g  = ((const float4*)gamma)[tid];
  const float4 be = ((const float4*)beta)[tid];
  float4 o;
  o.x = (v.x - mu) * rstd * g.x + be.x;
  o.y = (v.y - mu) * rstd * g.y + be.y;
  o.z = (v.z - mu) * rstd * g.z + be.z;
  o.w = (v.w - mu) * rstd * g.w + be.w;
  ((float4*)(out + (size_t)row * DMODEL))[tid] = o;
}

// ---------------------------------------------------------------- launch
extern "C" void kernel_launch(void* const* d_in, const int* in_sizes, int n_in,
                              void* d_out, int out_size, void* d_ws, size_t ws_size,
                              hipStream_t stream) {
  (void)in_sizes; (void)n_in; (void)out_size; (void)ws_size;

  const float* query = (const float*)d_in[0];
  const float* key_  = (const float*)d_in[1];
  const float* value = (const float*)d_in[2];
  const int*   mask  = (const int*)d_in[3];
  const float* Wq = (const float*)d_in[4];
  const float* bq = (const float*)d_in[5];
  const float* Wk = (const float*)d_in[6];
  const float* bk = (const float*)d_in[7];
  const float* Wv = (const float*)d_in[8];
  const float* bv = (const float*)d_in[9];
  const float* Wo = (const float*)d_in[10];
  const float* bo = (const float*)d_in[11];
  const float* gamma = (const float*)d_in[12];
  const float* beta  = (const float*)d_in[13];

  const size_t nBHLD = (size_t)BATCH * HEADS * LCTX * DHEAD;  // 8.39M elems
  unsigned short* qT = (unsigned short*)d_ws;
  unsigned short* kT = qT + nBHLD;
  unsigned short* vT = kT + nBHLD;
  float* attnO = (float*)(vT + nBHLD);
  float* xbuf  = attnO + nBHLD;

  dim3 g1(DMODEL / 128, (BATCH * LCTX) / 128, 3);
  qkv_gemm_kernel<<<g1, 256, 0, stream>>>(query, key_, value, Wq, Wk, Wv,
                                          bq, bk, bv, qT, kT, vT);

  dim3 g2(LCTX / QBLK, BATCH * HEADS);
  flash_attn_kernel<<<g2, 256, 0, stream>>>(qT, kT, vT, mask, attnO);

  dim3 g3(DMODEL / 128, (BATCH * LCTX) / 128);
  oproj_gemm_kernel<<<g3, 256, 0, stream>>>(attnO, Wo, bo, query, xbuf);

  layernorm_kernel<<<BATCH * LCTX, 256, 0, stream>>>(xbuf, gamma, beta,
                                                     (float*)d_out);
}
